// Attention_19963007992532
// MI455X (gfx1250) — matrix operand order repaired and checked by reading, not compiled, not run on hardware
//
#include <hip/hip_runtime.h>
#include <hip/hip_bf16.h>

#define SL 256
#define TL 256
#define BS 4
#define H  512

typedef __attribute__((ext_vector_type(16))) __bf16 v16bf;
typedef __attribute__((ext_vector_type(8)))  float  v8f;

// ---- fragment loaders matching CDNA5 16-bit WMMA VGPR layouts -------------
// A (16x32, bf16): lane L holds row M=L&15; elems 0..7 = K {kb+half*8 ..+7},
// elems 8..15 = K {kb+16+half*8 ..+7}  (half = L>>4)
// B (32x16, bf16): lane L holds col N=L&15; elems 0..7 = K {kb+half*8 ..+7},
// elems 8..15 = K {kb+16+half*8 ..+7}

__device__ __forceinline__ v16bf cvt_frag_f32(const float* p0, const float* p1) {
  float4 a0 = ((const float4*)p0)[0];
  float4 a1 = ((const float4*)p0)[1];
  float4 b0 = ((const float4*)p1)[0];
  float4 b1 = ((const float4*)p1)[1];
  v16bf r;
  r[0]=(__bf16)a0.x; r[1]=(__bf16)a0.y; r[2]=(__bf16)a0.z; r[3]=(__bf16)a0.w;
  r[4]=(__bf16)a1.x; r[5]=(__bf16)a1.y; r[6]=(__bf16)a1.z; r[7]=(__bf16)a1.w;
  r[8]=(__bf16)b0.x; r[9]=(__bf16)b0.y; r[10]=(__bf16)b0.z; r[11]=(__bf16)b0.w;
  r[12]=(__bf16)b1.x; r[13]=(__bf16)b1.y; r[14]=(__bf16)b1.z; r[15]=(__bf16)b1.w;
  return r;
}

__device__ __forceinline__ v16bf load_frag_bf(const __bf16* p0, const __bf16* p1) {
  v16bf r;
  #pragma unroll
  for (int i = 0; i < 8; ++i) { r[i] = p0[i]; r[i+8] = p1[i]; }
  return r;
}

__device__ __forceinline__ float fast_tanh(float x) {
  // tanh(x) = 1 - 2/(exp2(2*log2(e)*x) + 1); saturates correctly at +/-inf
  float e = __builtin_amdgcn_exp2f(x * 2.8853900817779268f);
  return 1.0f - 2.0f * __builtin_amdgcn_rcpf(e + 1.0f);
}

// ---- K1: e_proj = e @ We^T ; d_projb = d @ Wd^T + b  (WMMA bf16) ----------
__global__ void proj_kernel(const float* __restrict__ out_e,
                            const float* __restrict__ out_d,
                            const float* __restrict__ W,
                            const float* __restrict__ bias,
                            float* __restrict__ e_proj,
                            float* __restrict__ d_projb) {
  const int tid  = threadIdx.x;
  const int lane = tid & 31;
  const int wave = tid >> 5;
  const int half = lane >> 4;
  const int lm   = lane & 15;

  const int  bz  = blockIdx.z;
  const bool isD = bz >= BS;
  const int  b   = bz & (BS - 1);

  const float* src  = isD ? out_d   : out_e;
  float*       dst  = isD ? d_projb : e_proj;
  const int    woff = isD ? H : 0;

  const int m_base = blockIdx.x * 32 + (wave & 1) * 16;
  const int n_base = blockIdx.y * 64 + (wave >> 1) * 16;
  const int m = m_base + lm;           // seq position (A row)
  const int n = n_base + lm;           // output feature (B col)

  const float* arow = src + (size_t)m * (BS * H) + (size_t)b * H;   // A[m][h]
  const float* brow = W   + (size_t)n * (2 * H) + woff;             // B[h][n]=W[n][woff+h]

  v8f acc = {};
  #pragma unroll
  for (int kb = 0; kb < H; kb += 32) {
    v16bf a  = cvt_frag_f32(arow + kb + half * 8,  arow + kb + half * 8 + 16);
    v16bf bf = cvt_frag_f32(brow + kb + half * 8,  brow + kb + half * 8 + 16);
    acc = __builtin_amdgcn_wmma_f32_16x16x32_bf16(false, a, false, bf,
                                                  (short)0, acc, false, false);
  }
  const float bn = isD ? bias[n] : 0.0f;
  #pragma unroll
  for (int r = 0; r < 8; ++r) {
    int row = m_base + half * 8 + r;
    dst[(size_t)b * SL * H + (size_t)row * H + n] = acc[r] + bn;
  }
}

// ---- K0: eT[b][h][s] = bf16(out_e[s][b][h])  (for context-GEMM B frags) ---
__global__ void transpose_e_bf16(const float* __restrict__ out_e,
                                 __bf16* __restrict__ eT) {
  const int s  = threadIdx.x;          // 0..SL-1
  const int bh = blockIdx.x;           // b*H + h
  const int b  = bh >> 9;
  const int h  = bh & (H - 1);
  eT[(size_t)bh * SL + s] = (__bf16)out_e[(size_t)s * BS * H + (size_t)b * H + h];
}

// ---- K2a: scores[b][s][t] = sum_k v[k]*tanh(e_proj[s,k]+d_projb[t,k]) -----
// Double-buffered GLOBAL_LOAD_ASYNC_TO_LDS_B128 staging (ASYNCcnt-tracked),
// so the tanh/exp VALU stream overlaps the memory->LDS DMA completely.
#define KC   64
#define KSTR 68   // padded LDS row stride (floats): rows stay 16B aligned
#define NCH  (H / KC)

__global__ void scores_kernel(const float* __restrict__ e_proj,
                              const float* __restrict__ d_projb,
                              const float* __restrict__ v,
                              float* __restrict__ S) {
  extern __shared__ float sm[];
  float* epsB = sm;                       // [2][32][KSTR]
  float* dpsB = sm + 2 * 32 * KSTR;       // [2][32][KSTR]
  float* vsA  = sm + 4 * 32 * KSTR;       // [H]

  const int tid = threadIdx.x;
  const int b   = blockIdx.z;
  const int s0  = blockIdx.x * 32;
  const int t0  = blockIdx.y * 32;
  const int tl  = tid & 31;               // t within tile (lane-varying)
  const int sq  = tid >> 5;               // s group (4 s per thread)

  const float* eg = e_proj  + (size_t)b * SL * H + (size_t)s0 * H;
  const float* dg = d_projb + (size_t)b * TL * H + (size_t)t0 * H;

  const unsigned epsOff = (unsigned)(size_t)(void*)epsB;  // raw LDS byte offset
  const unsigned dpsOff = (unsigned)(size_t)(void*)dpsB;

  // stage full v vector into LDS once (visible after first barrier)
  for (int i = tid; i < H; i += 256) vsA[i] = v[i];

  // issue one K-chunk's tiles as async B128 copies (4 instrs/thread/chunk)
  auto issue = [&](int kb, int buf) {
    #pragma unroll
    for (int it = 0; it < 2; ++it) {
      int idx = tid + it * 256;           // float4 index within 32x64 tile
      int row = idx >> 4;                 // 16 float4 per row
      int c4  = (idx & 15) << 2;
      unsigned le = epsOff + (unsigned)(((buf * 32 + row) * KSTR + c4) * 4);
      unsigned ld = dpsOff + (unsigned)(((buf * 32 + row) * KSTR + c4) * 4);
      unsigned long long ge = (unsigned long long)(eg + (size_t)row * H + kb + c4);
      unsigned long long gd = (unsigned long long)(dg + (size_t)row * H + kb + c4);
      asm volatile("global_load_async_to_lds_b128 %0, %1, off"
                   :: "v"(le), "v"(ge) : "memory");
      asm volatile("global_load_async_to_lds_b128 %0, %1, off"
                   :: "v"(ld), "v"(gd) : "memory");
    }
  };

  float acc[4] = {0.f, 0.f, 0.f, 0.f};
  issue(0, 0);
  for (int c = 0; c < NCH; ++c) {
    const int buf = c & 1;
    if (c + 1 < NCH) {
      issue((c + 1) * KC, buf ^ 1);
      // async loads complete in order: <=4 outstanding => chunk c landed
      asm volatile("s_wait_asynccnt 0x4" ::: "memory");
    } else {
      asm volatile("s_wait_asynccnt 0x0" ::: "memory");
    }
    __syncthreads();

    const float* er = epsB + (size_t)(buf * 32 + sq * 4) * KSTR;  // broadcast rows
    const float* dr = dpsB + (size_t)(buf * 32 + tl) * KSTR;      // lane-varying
    const float* vr = vsA + c * KC;
    #pragma unroll 4
    for (int k = 0; k < KC; ++k) {
      float dv = dr[k];
      float vk = vr[k];
      #pragma unroll
      for (int j = 0; j < 4; ++j)
        acc[j] += vk * fast_tanh(er[(size_t)j * KSTR + k] + dv);
    }
    __syncthreads();   // all waves done with buf before it is overwritten
  }

  #pragma unroll
  for (int j = 0; j < 4; ++j) {
    int s = s0 + sq * 4 + j;
    S[(size_t)b * SL * TL + (size_t)s * TL + t0 + tl] = acc[j];
  }
}

// ---- K2b: attnT[b][t][s] = bf16(log_softmax over s of S[b][s][t]) ---------
__global__ void softmax_kernel(const float* __restrict__ S,
                               __bf16* __restrict__ attnT) {
  const int b = blockIdx.x;
  const int t = threadIdx.x;
  const float* col = S + (size_t)b * SL * TL + t;

  float m = -3.4e38f;
  for (int s = 0; s < SL; ++s) m = fmaxf(m, col[(size_t)s * TL]);
  float sum = 0.f;
  for (int s = 0; s < SL; ++s)
    sum += __builtin_amdgcn_exp2f((col[(size_t)s * TL] - m) * 1.4426950408889634f);
  float lse = m + 0.6931471805599453f * __builtin_amdgcn_logf(sum);

  __bf16* dst = attnT + (size_t)b * TL * SL + (size_t)t * SL;
  for (int s = 0; s < SL; ++s) dst[s] = (__bf16)(col[(size_t)s * TL] - lse);
}

// ---- K3: out[t][b][h] = sum_s attnT[b][t][s] * e[b][s][h]  (WMMA bf16) ----
__global__ void context_kernel(const __bf16* __restrict__ attnT,
                               const __bf16* __restrict__ eT,
                               float* __restrict__ out) {
  const int tid  = threadIdx.x;
  const int lane = tid & 31;
  const int wave = tid >> 5;
  const int half = lane >> 4;
  const int lm   = lane & 15;
  const int b    = blockIdx.z;

  const int m_base = blockIdx.x * 32 + (wave & 1) * 16;   // t tile
  const int n_base = blockIdx.y * 64 + (wave >> 1) * 16;  // h tile
  const int m = m_base + lm;       // t
  const int n = n_base + lm;       // h

  const __bf16* arow = attnT + (size_t)b * TL * SL + (size_t)m * SL;  // A[t][s]
  const __bf16* brow = eT    + (size_t)b * H  * SL + (size_t)n * SL;  // B[s][h]=eT[h][s]

  v8f acc = {};
  #pragma unroll
  for (int kb = 0; kb < SL; kb += 32) {
    v16bf a  = load_frag_bf(arow + kb + half * 8,  arow + kb + half * 8 + 16);
    v16bf bf = load_frag_bf(brow + kb + half * 8,  brow + kb + half * 8 + 16);
    acc = __builtin_amdgcn_wmma_f32_16x16x32_bf16(false, a, false, bf,
                                                  (short)0, acc, false, false);
  }
  #pragma unroll
  for (int r = 0; r < 8; ++r) {
    int row = m_base + half * 8 + r;                       // t
    out[(size_t)row * BS * H + (size_t)b * H + n] = acc[r];
  }
}

extern "C" void kernel_launch(void* const* d_in, const int* in_sizes, int n_in,
                              void* d_out, int out_size, void* d_ws, size_t ws_size,
                              hipStream_t stream) {
  (void)in_sizes; (void)n_in; (void)out_size; (void)ws_size;
  // inputs: 0=in_e (unused int), 1=out_e, 2=out_d, 3=W, 4=b, 5=v
  const float* out_e = (const float*)d_in[1];
  const float* out_d = (const float*)d_in[2];
  const float* W     = (const float*)d_in[3];
  const float* bias  = (const float*)d_in[4];
  const float* v     = (const float*)d_in[5];

  float* ws      = (float*)d_ws;
  float* e_proj  = ws;                                        // BS*SL*H f32
  float* d_projb = e_proj  + (size_t)BS * SL * H;             // BS*TL*H f32
  float* S       = d_projb + (size_t)BS * TL * H;             // BS*SL*TL f32
  __bf16* attnT  = (__bf16*)(S + (size_t)BS * SL * TL);       // BS*TL*SL bf16
  __bf16* eT     = attnT + (size_t)BS * TL * SL;              // BS*H*SL  bf16

  const size_t smem = (size_t)(4 * 32 * KSTR + H) * sizeof(float);  // 36 KB

  proj_kernel     <<<dim3(SL / 32, H / 64, 2 * BS), 256, 0, stream>>>(
      out_e, out_d, W, bias, e_proj, d_projb);
  transpose_e_bf16<<<dim3(BS * H), SL, 0, stream>>>(out_e, eT);
  scores_kernel   <<<dim3(SL / 32, TL / 32, BS), 256, smem, stream>>>(
      e_proj, d_projb, v, S);
  softmax_kernel  <<<dim3(BS), TL, 0, stream>>>(S, attnT);
  context_kernel  <<<dim3(TL / 32, H / 64, BS), 256, 0, stream>>>(
      attnT, eT, (float*)d_out);
}
